// Attention_12189117186326
// MI455X (gfx1250) — matrix operand order repaired and checked by reading, not compiled, not run on hardware
//
#include <hip/hip_runtime.h>
#include <hip/hip_bf16.h>

// ---------------------------------------------------------------------------
// Causal MHA (B=2, N=2048, DIM=1024, H=16, DH=64) for gfx1250 / MI455X.
// All matmuls in bf16 via v_wmma_f32_16x16x32_bf16 (wave32 WMMA).
// K/V tile staging in attn_out uses the Tensor Data Mover when available.
// ---------------------------------------------------------------------------

typedef __attribute__((ext_vector_type(16))) __bf16 v16bf;
typedef __attribute__((ext_vector_type(8)))  __bf16 v8bf;
typedef __attribute__((ext_vector_type(8)))  float  v8f;
typedef __attribute__((ext_vector_type(4)))  unsigned int u32x4;
typedef __attribute__((ext_vector_type(8)))  int  i32x8;
typedef __attribute__((ext_vector_type(4)))  int  i32x4;

#define B_     2
#define N_     2048
#define DIM_   1024
#define H_     16
#define DH_    64
#define INNER_ 1024
#define NEG_   (-3.0e38f)

#if __has_builtin(__builtin_amdgcn_tensor_load_to_lds)
#define HAVE_TDM 1
#else
#define HAVE_TDM 0
#endif

static __device__ inline v8f zero8() {
  v8f z = {0.f, 0.f, 0.f, 0.f, 0.f, 0.f, 0.f, 0.f};
  return z;
}

static __device__ inline v16bf combine16(v8bf lo, v8bf hi) {
  v16bf r;
#pragma unroll
  for (int e = 0; e < 8; ++e) { r[e] = lo[e]; r[e + 8] = hi[e]; }
  return r;
}

// A fragment (16x32 bf16) from a row-major [rows][ld] source.
// Lane m+16g holds row m, K = {8g..8g+7} in e0..7 and {16+8g..23+8g} in e8..15.
static __device__ inline v16bf load_a_frag(const __bf16* base, int ld, int row0, int k0) {
  int lane = threadIdx.x & 31;
  int m = lane & 15, g = lane >> 4;
  const __bf16* p = base + (size_t)(row0 + m) * ld + k0;
  v8bf lo = *(const v8bf*)(p + 8 * g);
  v8bf hi = *(const v8bf*)(p + 16 + 8 * g);
  return combine16(lo, hi);
}

// B fragment (32x16 bf16) from an N-major source: row n of `base` holds column n
// of the B matrix contiguously over K. Lane n+16g holds column n, K = {8g..8g+7} in e0..7 and {16+8g..23+8g} in e8..15.
static __device__ inline v16bf load_b_frag(const __bf16* base, int ld, int col0, int k0) {
  int lane = threadIdx.x & 31;
  int n = lane & 15, g = lane >> 4;
  const __bf16* p = base + (size_t)(col0 + n) * ld + k0;
  v8bf lo = *(const v8bf*)(p + 8 * g);
  v8bf hi = *(const v8bf*)(p + 16 + 8 * g);
  return combine16(lo, hi);
}

static __device__ inline v8f wmma_bf16(v16bf a, v16bf b, v8f c) {
  return __builtin_amdgcn_wmma_f32_16x16x32_bf16(false, a, false, b, (short)0, c,
                                                 false, false);
}

// ---------------------------------------------------------------------------
// Tensor Data Mover: 1-D contiguous copy of `nelem` bf16 elements from global
// memory into LDS at byte offset `lds_off`. D# layout per CDNA5 ISA:
//   group0: [1:0]=count=1, [63:32]=lds_addr, [120:64]=global_addr, [127:126]=2
//   group1: [17:16]=data_size(1->2B), [79:48]=tensor_dim0, [111:80]=tensor_dim1,
//           [127:112]=tile_dim0 ; dims 2..4 unused (groups 2/3 zero).
// ---------------------------------------------------------------------------
static __device__ inline void tdm_load_1d(unsigned lds_off, const void* gptr,
                                          unsigned nelem) {
#if HAVE_TDM
  unsigned long long ga = (unsigned long long)gptr;
  u32x4 g0 = {0u, 0u, 0u, 0u};
  g0[0] = 1u;                                    // count=1, user descriptor
  g0[1] = lds_off;                               // lds_addr
  g0[2] = (unsigned)ga;                          // global_addr lo
  g0[3] = (unsigned)(ga >> 32) | (2u << 30);     // global_addr hi | type=2
  i32x8 g1 = {0, 0, 0, 0, 0, 0, 0, 0};
  g1[0] = (int)(1u << 16);                       // data_size = 2 bytes
  g1[1] = (int)((nelem & 0xFFFFu) << 16);        // tensor_dim0[15:0]
  g1[2] = (int)(((nelem >> 16) & 0xFFFFu) | (1u << 16));  // dim0 hi | tensor_dim1=1
  g1[3] = (int)((nelem & 0xFFFFu) << 16);        // tile_dim0
  i32x4 gz = {0, 0, 0, 0};
#if __clang_major__ >= 23
  i32x8 gz8 = {0, 0, 0, 0, 0, 0, 0, 0};
  __builtin_amdgcn_tensor_load_to_lds(g0, g1, gz, gz, gz8, 0);
#else
  __builtin_amdgcn_tensor_load_to_lds(g0, g1, gz, gz, 0);
#endif
#endif
}

static __device__ inline void wait_tensorcnt0() {
#if __has_builtin(__builtin_amdgcn_s_wait_tensorcnt)
  __builtin_amdgcn_s_wait_tensorcnt((short)0);
#else
  asm volatile("s_wait_tensorcnt 0x0" ::: "memory");
#endif
}

// ---------------------------------------------------------------------------
// fp32 -> bf16 conversion kernels
// ---------------------------------------------------------------------------
__global__ void cvt_bf16(const float* __restrict__ src, __bf16* __restrict__ dst, int n) {
  int i = blockIdx.x * blockDim.x + threadIdx.x;
  int stride = gridDim.x * blockDim.x;
  for (; i < n; i += stride) dst[i] = (__bf16)src[i];
}

// src is [K][Ncols] row-major; dst is [Ncols][K] (N-major) bf16.
__global__ void cvt_bf16_t(const float* __restrict__ src, __bf16* __restrict__ dst,
                           int K, int Ncols) {
  int i = blockIdx.x * blockDim.x + threadIdx.x;
  int total = K * Ncols;
  int stride = gridDim.x * blockDim.x;
  for (; i < total; i += stride) {
    int k = i / Ncols, n = i - k * Ncols;
    dst[(size_t)n * K + k] = (__bf16)src[i];
  }
}

// ---------------------------------------------------------------------------
// QKV projection: qkv = x @ W_qkv. One wave computes a 16(M) x 64(N) tile.
// Epilogue scatters into q/k/v [B,H,N,DH] bf16, q pre-scaled by DH^-0.5 = 1/8.
// ---------------------------------------------------------------------------
__global__ void __launch_bounds__(32) gemm_qkv(const __bf16* __restrict__ xb,
                                               const __bf16* __restrict__ wT,
                                               __bf16* __restrict__ qb,
                                               __bf16* __restrict__ kb,
                                               __bf16* __restrict__ vb) {
  const int NTILES = (3 * INNER_) / 64;  // 48
  int mt = blockIdx.x / NTILES, nb = blockIdx.x % NTILES;
  int m0 = mt * 16, n0 = nb * 64;
  v8f acc[4] = {zero8(), zero8(), zero8(), zero8()};
  for (int k0 = 0; k0 < DIM_; k0 += 32) {
    if (k0 + 32 < DIM_)
      __builtin_prefetch((const void*)(xb + (size_t)m0 * DIM_ + k0 + 32), 0, 0);
    // Batch all fragment loads so a single wait covers the 4 wmmas.
    v16bf a  = load_a_frag(xb, DIM_, m0, k0);
    v16bf b0 = load_b_frag(wT, DIM_, n0 + 0,  k0);
    v16bf b1 = load_b_frag(wT, DIM_, n0 + 16, k0);
    v16bf b2 = load_b_frag(wT, DIM_, n0 + 32, k0);
    v16bf b3 = load_b_frag(wT, DIM_, n0 + 48, k0);
    acc[0] = wmma_bf16(a, b0, acc[0]);
    acc[1] = wmma_bf16(a, b1, acc[1]);
    acc[2] = wmma_bf16(a, b2, acc[2]);
    acc[3] = wmma_bf16(a, b3, acc[3]);
  }
  int lane = threadIdx.x & 31, n = lane & 15, g = lane >> 4;
#pragma unroll
  for (int t = 0; t < 4; ++t) {
    int c = n0 + t * 16 + n;
    int which = c >> 10;            // 0=q 1=k 2=v
    int hc = (c >> 6) & 15;         // head
    int d = c & 63;                 // dim within head
    __bf16* dst = which == 0 ? qb : (which == 1 ? kb : vb);
    float scl = which == 0 ? 0.125f : 1.0f;
#pragma unroll
    for (int r = 0; r < 8; ++r) {
      int row = m0 + r + 8 * g;
      int bb = row >> 11, nn = row & 2047;
      dst[((size_t)((bb * H_ + hc) * N_) + nn) * DH_ + d] = (__bf16)(acc[t][r] * scl);
    }
  }
}

// ---------------------------------------------------------------------------
// Pass 1: per-row softmax stats (running max m, sum-of-exp l), flash style.
// One wave per (b, h, 16-row i-tile). Causal + key-padding mask + bias.
// ---------------------------------------------------------------------------
__global__ void __launch_bounds__(32) attn_stats(const __bf16* __restrict__ qb,
                                                 const __bf16* __restrict__ kb,
                                                 const unsigned char* __restrict__ mask,
                                                 const float* __restrict__ bias,
                                                 float* __restrict__ mrow,
                                                 float* __restrict__ lrow) {
  int blk = blockIdx.x;
  int it = blk & 127, h = (blk >> 7) & 15, b = blk >> 11;
  int i0 = it * 16;
  const size_t hbase = (size_t)((b * H_ + h) * N_);
  const __bf16* qt = qb + hbase * DH_;
  const __bf16* kt = kb + hbase * DH_;
  v16bf aq0 = load_a_frag(qt, DH_, i0, 0);
  v16bf aq1 = load_a_frag(qt, DH_, i0, 32);
  int lane = threadIdx.x & 31, n = lane & 15, g = lane >> 4;
  float rm[8], rl[8];
#pragma unroll
  for (int r = 0; r < 8; ++r) { rm[r] = NEG_; rl[r] = 0.f; }
  const float* biash = bias + (size_t)h * N_ * N_;
  for (int jt = 0; jt <= it; ++jt) {
    int j0 = jt * 16;
    v8f s = zero8();
    s = wmma_bf16(aq0, load_b_frag(kt, DH_, j0, 0), s);
    s = wmma_bf16(aq1, load_b_frag(kt, DH_, j0, 32), s);
    int jg = j0 + n;
    bool jm = mask[b * N_ + jg] != 0;
#pragma unroll
    for (int r = 0; r < 8; ++r) {
      int ig = i0 + r + 8 * g;
      float x = s[r];
      if (jm || jg > ig) x = NEG_;
      x += biash[(size_t)ig * N_ + jg];
      // row of the tile lives across 16 lanes (same g); reduce max then sum.
      float tm = x;
      tm = fmaxf(tm, __shfl_xor(tm, 1, 16));
      tm = fmaxf(tm, __shfl_xor(tm, 2, 16));
      tm = fmaxf(tm, __shfl_xor(tm, 4, 16));
      tm = fmaxf(tm, __shfl_xor(tm, 8, 16));
      float mnew = fmaxf(rm[r], tm);
      float p = __expf(x - mnew);
      p += __shfl_xor(p, 1, 16);
      p += __shfl_xor(p, 2, 16);
      p += __shfl_xor(p, 4, 16);
      p += __shfl_xor(p, 8, 16);
      rl[r] = rl[r] * __expf(rm[r] - mnew) + p;
      rm[r] = mnew;
    }
  }
  if (n == 0) {
#pragma unroll
    for (int r = 0; r < 8; ++r) {
      size_t off = hbase + i0 + r + 8 * g;
      mrow[off] = rm[r];
      lrow[off] = rl[r];
    }
  }
}

// ---------------------------------------------------------------------------
// Pass 2a: out_head = softmax(scores) @ V per (b, h, i-tile).
// K/V 32x64 tiles are staged into LDS via the Tensor Data Mover (contiguous
// 4 KB each), then score tiles are recomputed, p = exp(s-m)/l -> bf16 in LDS,
// and attn x V runs via WMMA (V transposed in LDS for contiguous B columns).
// ---------------------------------------------------------------------------
__global__ void __launch_bounds__(32) attn_out(const __bf16* __restrict__ qb,
                                               const __bf16* __restrict__ kb,
                                               const __bf16* __restrict__ vb,
                                               const unsigned char* __restrict__ mask,
                                               const float* __restrict__ bias,
                                               const float* __restrict__ mrow,
                                               const float* __restrict__ lrow,
                                               __bf16* __restrict__ ho) {
  __shared__ __attribute__((aligned(16))) __bf16 kTile[32 * 64];  // [j][d]
  __shared__ __attribute__((aligned(16))) __bf16 vTile[32 * 64];  // [j][d]
  __shared__ __attribute__((aligned(16))) __bf16 attnT[16 * 32];  // [m][j]
  __shared__ __attribute__((aligned(16))) __bf16 vT[64 * 40];     // [d][j], ld=40
  int blk = blockIdx.x;
  int it = blk & 127, h = (blk >> 7) & 15, b = blk >> 11;
  int i0 = it * 16;
  const size_t hbase = (size_t)((b * H_ + h) * N_);
  const __bf16* qt = qb + hbase * DH_;
  const __bf16* kt = kb + hbase * DH_;
  const __bf16* vt = vb + hbase * DH_;
  v16bf aq0 = load_a_frag(qt, DH_, i0, 0);
  v16bf aq1 = load_a_frag(qt, DH_, i0, 32);
  int lane = threadIdx.x & 31, n = lane & 15, g = lane >> 4;
  float rm[8], rli[8];
#pragma unroll
  for (int r = 0; r < 8; ++r) {
    size_t off = hbase + i0 + r + 8 * g;
    rm[r] = mrow[off];
    rli[r] = 1.0f / lrow[off];
  }
  const float* biash = bias + (size_t)h * N_ * N_;
  unsigned kOff = (unsigned)(uintptr_t)(void*)kTile;
  unsigned vOff = (unsigned)(uintptr_t)(void*)vTile;
  v8f acc[4] = {zero8(), zero8(), zero8(), zero8()};
  for (int j0 = 0; j0 <= i0 + 15; j0 += 32) {
#if HAVE_TDM
    // Drain pending LDS reads of the previous tile before the TDM overwrites it
    // (TDM traffic is unordered vs DS traffic).
    asm volatile("s_wait_dscnt 0x0" ::: "memory");
    tdm_load_1d(kOff, (const void*)(kt + (size_t)j0 * DH_), 32 * DH_);
    tdm_load_1d(vOff, (const void*)(vt + (size_t)j0 * DH_), 32 * DH_);
    wait_tensorcnt0();
#else
    {  // Fallback: cooperative global->LDS staging, lane owns row j0+lane.
      const __bf16* krow = kt + (size_t)(j0 + lane) * DH_;
      const __bf16* vrow = vt + (size_t)(j0 + lane) * DH_;
#pragma unroll
      for (int c = 0; c < 8; ++c) {
        *(v8bf*)(kTile + lane * 64 + c * 8) = *(const v8bf*)(krow + c * 8);
        *(v8bf*)(vTile + lane * 64 + c * 8) = *(const v8bf*)(vrow + c * 8);
      }
    }
#endif
    // Transpose V tile: vT[d][j] = vTile[j][d]; lane owns row j = lane.
#pragma unroll
    for (int c = 0; c < 8; ++c) {
      v8bf vv = *(const v8bf*)(vTile + lane * 64 + c * 8);
#pragma unroll
      for (int e = 0; e < 8; ++e) vT[(c * 8 + e) * 40 + lane] = vv[e];
    }
#pragma unroll
    for (int sub = 0; sub < 2; ++sub) {
      int js = j0 + sub * 16;
      v8f s = zero8();
      s = wmma_bf16(aq0, load_b_frag(kTile, DH_, sub * 16, 0), s);
      s = wmma_bf16(aq1, load_b_frag(kTile, DH_, sub * 16, 32), s);
      int jg = js + n;
      bool jm = mask[b * N_ + jg] != 0;
#pragma unroll
      for (int r = 0; r < 8; ++r) {
        int ig = i0 + r + 8 * g;
        float x = s[r];
        if (jm || jg > ig) x = NEG_;
        x += biash[(size_t)ig * N_ + jg];
        float p = __expf(x - rm[r]) * rli[r];
        attnT[(r + 8 * g) * 32 + sub * 16 + n] = (__bf16)p;
      }
    }
    // Same-wave LDS ops are in-order (DScnt) -> stores visible to loads below.
    v16bf af = load_a_frag(attnT, 32, 0, 0);
#pragma unroll
    for (int dt = 0; dt < 4; ++dt) {
      v16bf bv = load_b_frag(vT, 40, dt * 16, 0);
      acc[dt] = wmma_bf16(af, bv, acc[dt]);
    }
  }
  // Write head output as bf16 [B*N][INNER] with col = h*64 + d.
#pragma unroll
  for (int dt = 0; dt < 4; ++dt) {
    int col = h * DH_ + dt * 16 + n;
#pragma unroll
    for (int r = 0; r < 8; ++r) {
      int row = b * N_ + i0 + r + 8 * g;
      ho[(size_t)row * INNER_ + col] = (__bf16)acc[dt][r];
    }
  }
}

// ---------------------------------------------------------------------------
// Pass 2b: attn_avg[b,i,j] = mean_h exp(s-m)/l. One wave per (b, i-tile, j-tile)
// looping all 16 heads. Upper-triangle tiles are exactly zero.
// ---------------------------------------------------------------------------
__global__ void __launch_bounds__(32) attn_mean(const __bf16* __restrict__ qb,
                                                const __bf16* __restrict__ kb,
                                                const unsigned char* __restrict__ mask,
                                                const float* __restrict__ bias,
                                                const float* __restrict__ mrow,
                                                const float* __restrict__ lrow,
                                                float* __restrict__ avg) {
  int blk = blockIdx.x;
  int jt = blk & 127, it = (blk >> 7) & 127, b = blk >> 14;
  int i0 = it * 16, j0 = jt * 16;
  int lane = threadIdx.x & 31, n = lane & 15, g = lane >> 4;
  float* avgb = avg + (size_t)b * N_ * N_;
  if (jt > it) {  // causal: whole tile is zero (d_out is poisoned, must write)
#pragma unroll
    for (int r = 0; r < 8; ++r)
      avgb[(size_t)(i0 + r + 8 * g) * N_ + j0 + n] = 0.0f;
    return;
  }
  int jg = j0 + n;
  bool jm = mask[b * N_ + jg] != 0;
  float aacc[8];
#pragma unroll
  for (int r = 0; r < 8; ++r) aacc[r] = 0.f;
  for (int h = 0; h < H_; ++h) {
    const size_t hbase = (size_t)((b * H_ + h) * N_);
    const __bf16* qt = qb + hbase * DH_;
    const __bf16* kt = kb + hbase * DH_;
    v8f s = zero8();
    s = wmma_bf16(load_a_frag(qt, DH_, i0, 0), load_b_frag(kt, DH_, j0, 0), s);
    s = wmma_bf16(load_a_frag(qt, DH_, i0, 32), load_b_frag(kt, DH_, j0, 32), s);
    const float* biash = bias + (size_t)h * N_ * N_;
#pragma unroll
    for (int r = 0; r < 8; ++r) {
      int ig = i0 + r + 8 * g;
      float x = s[r];
      if (jm || jg > ig) x = NEG_;
      x += biash[(size_t)ig * N_ + jg];
      size_t off = hbase + ig;
      aacc[r] += __expf(x - mrow[off]) / lrow[off];
    }
  }
#pragma unroll
  for (int r = 0; r < 8; ++r)
    avgb[(size_t)(i0 + r + 8 * g) * N_ + j0 + n] = aacc[r] * (1.0f / 16.0f);
}

// ---------------------------------------------------------------------------
// Output projection: out = head_out @ W_out + b_out (f32 result).
// ---------------------------------------------------------------------------
__global__ void __launch_bounds__(32) gemm_out(const __bf16* __restrict__ ho,
                                               const __bf16* __restrict__ wT,
                                               const float* __restrict__ bout,
                                               float* __restrict__ out) {
  int mt = blockIdx.x >> 4, nb = blockIdx.x & 15;
  int m0 = mt * 16, n0 = nb * 64;
  v8f acc[4] = {zero8(), zero8(), zero8(), zero8()};
  for (int k0 = 0; k0 < INNER_; k0 += 32) {
    if (k0 + 32 < INNER_)
      __builtin_prefetch((const void*)(ho + (size_t)m0 * INNER_ + k0 + 32), 0, 0);
    v16bf a  = load_a_frag(ho, INNER_, m0, k0);
    v16bf b0 = load_b_frag(wT, INNER_, n0 + 0,  k0);
    v16bf b1 = load_b_frag(wT, INNER_, n0 + 16, k0);
    v16bf b2 = load_b_frag(wT, INNER_, n0 + 32, k0);
    v16bf b3 = load_b_frag(wT, INNER_, n0 + 48, k0);
    acc[0] = wmma_bf16(a, b0, acc[0]);
    acc[1] = wmma_bf16(a, b1, acc[1]);
    acc[2] = wmma_bf16(a, b2, acc[2]);
    acc[3] = wmma_bf16(a, b3, acc[3]);
  }
  int lane = threadIdx.x & 31, n = lane & 15, g = lane >> 4;
#pragma unroll
  for (int t = 0; t < 4; ++t) {
    int c = n0 + t * 16 + n;
    float bb = bout[c];
#pragma unroll
    for (int r = 0; r < 8; ++r) {
      int row = m0 + r + 8 * g;
      out[(size_t)row * DIM_ + c] = acc[t][r] + bb;
    }
  }
}

// ---------------------------------------------------------------------------
extern "C" void kernel_launch(void* const* d_in, const int* in_sizes, int n_in,
                              void* d_out, int out_size, void* d_ws, size_t ws_size,
                              hipStream_t stream) {
  const float* x            = (const float*)d_in[0];
  const unsigned char* mask = (const unsigned char*)d_in[1];  // jax bool = 1 byte
  const float* bias         = (const float*)d_in[2];          // [1,H,N,N]
  const float* Wqkv         = (const float*)d_in[3];          // [DIM, 3*INNER]
  const float* Wout         = (const float*)d_in[4];          // [INNER, DIM]
  const float* bout         = (const float*)d_in[5];          // [DIM]
  float* out = (float*)d_out;                       // [B,N,DIM]
  float* avg = out + (size_t)B_ * N_ * DIM_;        // [B,N,N]

  char* w = (char*)d_ws;
  auto alloc = [&](size_t bytes) -> char* {
    char* p = w;
    w += (bytes + 255) & ~(size_t)255;
    return p;
  };
  __bf16* xb    = (__bf16*)alloc((size_t)B_ * N_ * DIM_ * 2);
  __bf16* wqkvT = (__bf16*)alloc((size_t)3 * INNER_ * DIM_ * 2);
  __bf16* woutT = (__bf16*)alloc((size_t)DIM_ * INNER_ * 2);
  __bf16* qb    = (__bf16*)alloc((size_t)B_ * H_ * N_ * DH_ * 2);
  __bf16* kb    = (__bf16*)alloc((size_t)B_ * H_ * N_ * DH_ * 2);
  __bf16* vb    = (__bf16*)alloc((size_t)B_ * H_ * N_ * DH_ * 2);
  float*  mrow  = (float*)alloc((size_t)B_ * H_ * N_ * 4);
  float*  lrow  = (float*)alloc((size_t)B_ * H_ * N_ * 4);
  __bf16* ho    = (__bf16*)alloc((size_t)B_ * N_ * INNER_ * 2);

  cvt_bf16<<<1024, 256, 0, stream>>>(x, xb, B_ * N_ * DIM_);
  cvt_bf16_t<<<1024, 256, 0, stream>>>(Wqkv, wqkvT, DIM_, 3 * INNER_);
  cvt_bf16_t<<<1024, 256, 0, stream>>>(Wout, woutT, INNER_, DIM_);

  gemm_qkv<<<(B_ * N_ / 16) * (3 * INNER_ / 64), 32, 0, stream>>>(xb, wqkvT, qb, kb, vb);

  attn_stats<<<B_ * H_ * (N_ / 16), 32, 0, stream>>>(qb, kb, mask, bias, mrow, lrow);
  attn_out<<<B_ * H_ * (N_ / 16), 32, 0, stream>>>(qb, kb, vb, mask, bias, mrow, lrow, ho);
  attn_mean<<<B_ * (N_ / 16) * (N_ / 16), 32, 0, stream>>>(qb, kb, mask, bias, mrow, lrow, avg);

  gemm_out<<<(B_ * N_ / 16) * (DIM_ / 64), 32, 0, stream>>>(ho, woutT, bout, out);
}
